// NaiveGRUModel_51436528337763
// MI455X (gfx1250) — matrix operand order repaired and checked by reading, not compiled, not run on hardware
//
#include <hip/hip_runtime.h>
#include <hip/hip_bf16.h>
#include <stdint.h>

// ---------------------------------------------------------------------------
// GRU (T=8192, H=512, L=5) + FC(512->256) for MI455X / gfx1250 (wave32, WMMA)
//
//   per layer l:
//     Gz/Gr/Gh[T,H] = Xbf16 @ W{z,r,h}^T   (bf16 WMMA 16x16x32, f32 acc,
//                                           B tile staged via async->LDS,
//                                           double buffered)
//     sequential recurrence over t: U{z,r,h} rows LDS-resident via TDM
//     tensor_load_to_lds (192KB of the 320KB/WGP), 16 cooperating WGs,
//     2 global sense barriers per step, h / r*h broadcast via async->LDS.
//   out[T,O] = Ybf16 @ fcW^T + fcb
//   h_final[L,H] appended at d_out + T*O
// ---------------------------------------------------------------------------

#define T_FRAMES 8192
#define HID      512
#define OUTF     256
#define NLAYER   5
#define NWG      16
#define RPW      (HID / NWG)   // 32 rows of each U matrix per workgroup
#define SBSTRIDE 40            // bf16 elems per LDS B row (32 + 8 pad = 80B)

typedef __bf16        bf16x16 __attribute__((ext_vector_type(16)));
typedef float         f32x8   __attribute__((ext_vector_type(8)));
typedef float         f32x4   __attribute__((ext_vector_type(4)));
typedef unsigned int  u32x4   __attribute__((ext_vector_type(4)));
typedef unsigned int  u32x8   __attribute__((ext_vector_type(8)));

union Frag { bf16x16 v; u32x4 q[2]; };

// ---------------------------------------------------------------------------
// CDNA5 async / TDM helpers (inline asm per CDNA5 ISA)
// ---------------------------------------------------------------------------
__device__ inline unsigned lds_off_of(const void* p) {
  // generic LDS pointer: addr[31:0] is the wave-relative LDS offset (ISA 10.2)
  return (unsigned)(uintptr_t)p;
}

__device__ inline void async_load_b128(unsigned lds_byte_off, const void* g) {
  // GLOBAL_LOAD_ASYNC_TO_LDS_B128, GV mode; tracked by ASYNCcnt
  asm volatile("global_load_async_to_lds_b128 %0, %1, off"
               :: "v"(lds_byte_off), "v"((unsigned long long)(uintptr_t)g)
               : "memory");
}

__device__ inline void wait_async0() {
  asm volatile("s_wait_asynccnt 0x0" ::: "memory");
}

// 2D TDM load: tile (tile_d1 rows x tile_d0 elems) of a row-major tensor
// with row stride stride0 (elems), element size 4B, into LDS at lds_byte_off.
// Descriptor layout per ISA 8.3/8.4 (groups 0,1; groups 2,3 NULL => 2D).
__device__ inline void tdm_load_2d_f32(unsigned lds_byte_off, const void* g,
                                       unsigned tensor_d0, unsigned tensor_d1,
                                       unsigned tile_d0, unsigned tile_d1,
                                       unsigned stride0) {
  const unsigned long long ga = (unsigned long long)(uintptr_t)g;
  u32x4 g0;
  g0[0] = 1u;                                   // count=1, user descriptor
  g0[1] = lds_byte_off;                         // lds_addr [63:32]
  g0[2] = (unsigned)ga;                         // global_addr [95:64]
  g0[3] = (unsigned)((ga >> 32) & 0x01FFFFFFull) | (2u << 30); // addr hi, type=2
  u32x8 g1;
  g1[0] = (2u << 16);                           // wg_mask=0, data_size=2 (4B)
  g1[1] = (tensor_d0 & 0xFFFFu) << 16;          // tensor_dim0[15:0] @ bits63:48
  g1[2] = ((tensor_d0 >> 16) & 0xFFFFu) | ((tensor_d1 & 0xFFFFu) << 16);
  g1[3] = ((tensor_d1 >> 16) & 0xFFFFu) | ((tile_d0 & 0xFFFFu) << 16);
  g1[4] = (tile_d1 & 0xFFFFu);                  // tile_dim1, tile_dim2=0
  g1[5] = stride0;                              // tensor_dim0_stride[31:0]
  g1[6] = 0u;                                   // stride0 hi, dim1_stride lo
  g1[7] = 0u;
  asm volatile("tensor_load_to_lds %0, %1" :: "s"(g0), "s"(g1) : "memory");
}

// ---------------------------------------------------------------------------
// fp32 -> bf16 conversion (grid-stride)
// ---------------------------------------------------------------------------
__global__ void cvt_f32_bf16(const float* __restrict__ in,
                             __hip_bfloat16* __restrict__ out, int n) {
  for (int i = blockIdx.x * blockDim.x + threadIdx.x; i < n;
       i += gridDim.x * blockDim.x)
    out[i] = __float2bfloat16(in[i]);
}

__global__ void init_counter(unsigned* c) { *c = 0u; }

// ---------------------------------------------------------------------------
// C[M,N](f32) = A[M,K](bf16) @ B[N,K](bf16)^T (+bias), WMMA bf16 16x16x32.
// Block: 256 thr = 8 waves; wave tile 16(M) x 64(N); block tile 128 x 64.
// B tile (64 rows x 32 k) staged global->LDS with async b128 loads,
// double buffered; each thread stages one 16B segment per k-step.
// A frag (16x32): lane m=lane%16, khalf=lane/16; 2x b128 from global.
// B frag (32x16): row n0+n of W, K=khalf*8+{0..7} and 16+khalf*8+{0..7}; 2x ds_load_b128.
// D (f32 16x16): vgpr r -> row r + (lane/16)*8, col lane%16.
// ---------------------------------------------------------------------------
__global__ __launch_bounds__(256)
void wmma_gemm_xwT(const __hip_bfloat16* __restrict__ A,
                   const __hip_bfloat16* __restrict__ B,
                   const float* __restrict__ bias,
                   float* __restrict__ C, int M, int N, int K) {
  __shared__ __hip_bfloat16 sB[2][64 * SBSTRIDE];

  const int tid   = threadIdx.x;
  const int lane  = tid & 31;
  const int wave  = tid >> 5;
  const int m0    = blockIdx.x * 128 + wave * 16;
  const int n0    = blockIdx.y * 64;
  const int l16   = lane & 15;
  const int khalf = lane >> 4;

  const int brow = tid >> 2, bseg = tid & 3;   // this thread's staging slot
  const unsigned sBoff[2] = {lds_off_of(&sB[0][0]), lds_off_of(&sB[1][0])};
  const unsigned dstoff = (unsigned)(brow * SBSTRIDE + bseg * 8) * 2u;
  const __hip_bfloat16* brp = B + (size_t)(n0 + brow) * K + bseg * 8;

  async_load_b128(sBoff[0] + dstoff, brp);     // prime k-tile 0

  f32x8 acc[4] = {};
  const int KT = K >> 5;
  for (int kt = 0; kt < KT; ++kt) {
    const int k0 = kt << 5;
    wait_async0();                 // this wave's fill of sB[kt&1] done
    __syncthreads();               // everyone's fill done / prev reads done
    if (kt + 1 < KT)
      async_load_b128(sBoff[(kt + 1) & 1] + dstoff, brp + k0 + 32);

    Frag a;
    const __hip_bfloat16* ap = A + (size_t)(m0 + l16) * K + (k0 + khalf * 8);
    a.q[0] = *reinterpret_cast<const u32x4*>(ap);
    a.q[1] = *reinterpret_cast<const u32x4*>(ap + 16);
    __builtin_prefetch(ap + 32, 0, 0);         // global_prefetch next k-tile

    const __hip_bfloat16* bb = sB[kt & 1];
#pragma unroll
    for (int nt = 0; nt < 4; ++nt) {
      Frag b;
      const __hip_bfloat16* bp = bb + (nt * 16 + l16) * SBSTRIDE + khalf * 8;
      b.q[0] = *reinterpret_cast<const u32x4*>(bp);
      b.q[1] = *reinterpret_cast<const u32x4*>(bp + 16);
      acc[nt] = __builtin_amdgcn_wmma_f32_16x16x32_bf16(
          false, a.v, false, b.v, (short)0, acc[nt], false, false);
    }
  }
  const int crow = khalf * 8;
#pragma unroll
  for (int nt = 0; nt < 4; ++nt) {
    const int col = n0 + nt * 16 + l16;
    const float bv = bias ? bias[col] : 0.f;
#pragma unroll
    for (int r = 0; r < 8; ++r)
      C[(size_t)(m0 + crow + r) * N + col] = acc[nt][r] + bv;
  }
}

// ---------------------------------------------------------------------------
// Global sense barrier across the NWG persistent workgroups (monotonic count).
// ---------------------------------------------------------------------------
__device__ inline void grid_barrier(unsigned* cnt, unsigned target) {
  __syncthreads();
  if (threadIdx.x == 0) {
    __threadfence();                       // release our global stores
    atomicAdd(cnt, 1u);
    while (__hip_atomic_load(cnt, __ATOMIC_ACQUIRE,
                             __HIP_MEMORY_SCOPE_AGENT) < target)
      __builtin_amdgcn_s_sleep(1);
  }
  __syncthreads();
  __threadfence();                         // acquire: invalidate stale WGP$
}

// ---------------------------------------------------------------------------
// Recurrence for one layer. NWG blocks x 256 threads, persistent over T steps.
// Each block owns rows [wg*32, wg*32+32) of Uz/Ur/Uh, staged fp32 into LDS
// by three TDM tensor_load_to_lds descriptors (3*32*512*4 = 192KB of the
// 320KB/WGP). Full h / r*h vectors (2KB) exchanged through L2 with async->LDS
// loads and two global barriers per step.
// ---------------------------------------------------------------------------
__global__ __launch_bounds__(256)
void gru_layer_recurrence(
    const float* __restrict__ Uz, const float* __restrict__ Ur,
    const float* __restrict__ Uh, const float* __restrict__ Gz,
    const float* __restrict__ Gr, const float* __restrict__ Gh,
    const float* __restrict__ bWz, const float* __restrict__ bUz,
    const float* __restrict__ bWr, const float* __restrict__ bUr,
    const float* __restrict__ bWh, const float* __restrict__ bUh,
    float* __restrict__ hbuf, float* __restrict__ rhbuf,
    unsigned* __restrict__ counter, float* __restrict__ Y,
    float* __restrict__ hfinal, int layer, int T) {
  extern __shared__ float smem[];
  float* sU   = smem;                   // [3*RPW*HID] U rows (z|r|h)
  float* sh   = sU + 3 * RPW * HID;     // [HID] full h_{t-1}
  float* srh  = sh + HID;               // [HID] full r*h
  float* zbuf = srh + HID;              // [RPW]
  float* htil = zbuf + RPW;             // [RPW]
  float* bsum = htil + RPW;             // [3*RPW] combined biases

  const int tid = threadIdx.x;
  const int wg  = blockIdx.x;

  // ---- stage this WG's U rows with the Tensor Data Mover (one 2D tile/gate)
  if (tid < 32) {
    const size_t rowbase = (size_t)wg * RPW * HID;
    tdm_load_2d_f32(lds_off_of(sU),                 Uz + rowbase,
                    HID, HID, HID, RPW, HID);
    tdm_load_2d_f32(lds_off_of(sU + RPW * HID),     Ur + rowbase,
                    HID, HID, HID, RPW, HID);
    tdm_load_2d_f32(lds_off_of(sU + 2 * RPW * HID), Uh + rowbase,
                    HID, HID, HID, RPW, HID);
    __builtin_amdgcn_s_wait_tensorcnt(0);
  }
  for (int i = tid; i < HID; i += 256) sh[i] = 0.f;   // h_0 = 0
  if (tid < RPW) {
    const int col = wg * RPW + tid;
    bsum[tid]           = bWz[col] + bUz[col];
    bsum[RPW + tid]     = bWr[col] + bUr[col];
    bsum[2 * RPW + tid] = bWh[col] + bUh[col];
  }
  __syncthreads();

  const unsigned sh_off  = lds_off_of(sh);
  const unsigned srh_off = lds_off_of(srh);
  unsigned bar = (unsigned)layer * (unsigned)T * 2u;

  for (int t = 0; t < T; ++t) {
    // ---- phase 1: z (gate 0) and r (gate 1): 4 lanes x 128 K per row ----
    {
      const int task = tid >> 2, part = tid & 3;
      const int row = task & (RPW - 1), g = task >> 5;
      const f32x4* u4 = reinterpret_cast<const f32x4*>(
          &sU[(g * RPW + row) * HID + part * 128]);
      const f32x4* h4 = reinterpret_cast<const f32x4*>(&sh[part * 128]);
      float s = 0.f;
#pragma unroll
      for (int k = 0; k < 32; ++k) {
        const f32x4 a = u4[k], b = h4[k];
        s += a[0] * b[0] + a[1] * b[1] + a[2] * b[2] + a[3] * b[3];
      }
      s += __shfl_xor(s, 1);
      s += __shfl_xor(s, 2);
      if (part == 0) {
        const int col = wg * RPW + row;
        const float* G = (g == 0) ? Gz : Gr;
        const float pre = G[(size_t)t * HID + col] + bsum[g * RPW + row] + s;
        const float sig = 1.f / (1.f + __expf(-pre));
        if (g == 0) zbuf[row] = sig;
        else        rhbuf[col] = sig * sh[col];   // broadcast r*h slice
      }
    }
    grid_barrier(counter, (++bar) * NWG);
    // pull full r*h vector into LDS via async engine (128 x b128 = 2KB)
    if (tid < 128)
      async_load_b128(srh_off + (unsigned)tid * 16u, rhbuf + tid * 4);
    wait_async0();
    __syncthreads();

    // ---- phase 2: candidate h~: 8 lanes x 64 K per row ----
    {
      const int row = tid >> 3, part = tid & 7;
      const f32x4* u4 = reinterpret_cast<const f32x4*>(
          &sU[(2 * RPW + row) * HID + part * 64]);
      const f32x4* r4 = reinterpret_cast<const f32x4*>(&srh[part * 64]);
      float s = 0.f;
#pragma unroll
      for (int k = 0; k < 16; ++k) {
        const f32x4 a = u4[k], b = r4[k];
        s += a[0] * b[0] + a[1] * b[1] + a[2] * b[2] + a[3] * b[3];
      }
      s += __shfl_xor(s, 1);
      s += __shfl_xor(s, 2);
      s += __shfl_xor(s, 4);
      if (part == 0) {
        const int col = wg * RPW + row;
        htil[row] = tanhf(Gh[(size_t)t * HID + col] + bsum[2 * RPW + row] + s);
      }
    }
    __syncthreads();
    if (tid < RPW) {
      const int col = wg * RPW + tid;
      const float z = zbuf[tid];
      const float hn = (1.f - z) * sh[col] + z * htil[tid];
      hbuf[col] = hn;                       // broadcast new-h slice
      Y[(size_t)t * HID + col] = hn;        // layer output sequence
      if (t == T - 1) hfinal[col] = hn;     // h_final[l]
    }
    grid_barrier(counter, (++bar) * NWG);
    // pull full h_t vector into LDS via async engine
    if (tid < 128)
      async_load_b128(sh_off + (unsigned)tid * 16u, hbuf + tid * 4);
    wait_async0();
    __syncthreads();
  }
}

// ---------------------------------------------------------------------------
extern "C" void kernel_launch(void* const* d_in, const int* in_sizes, int n_in,
                              void* d_out, int out_size, void* d_ws,
                              size_t ws_size, hipStream_t stream) {
  (void)in_sizes; (void)n_in; (void)out_size; (void)ws_size;
  const float* x   = (const float*)d_in[0];
  const float* Wz  = (const float*)d_in[1];
  const float* bWz = (const float*)d_in[2];
  const float* Uz  = (const float*)d_in[3];
  const float* bUz = (const float*)d_in[4];
  const float* Wr  = (const float*)d_in[5];
  const float* bWr = (const float*)d_in[6];
  const float* Ur  = (const float*)d_in[7];
  const float* bUr = (const float*)d_in[8];
  const float* Wh  = (const float*)d_in[9];
  const float* bWh = (const float*)d_in[10];
  const float* Uh  = (const float*)d_in[11];
  const float* bUh = (const float*)d_in[12];
  const float* fcW = (const float*)d_in[13];
  const float* fcb = (const float*)d_in[14];
  float* out = (float*)d_out;

  // workspace carve-up (256B aligned slices)
  char* base = (char*)d_ws;
  size_t off = 0;
  auto take = [&](size_t bytes) -> char* {
    char* p = base + off;
    off = (off + bytes + 255) & ~(size_t)255;
    return p;
  };
  unsigned* counter     = (unsigned*)take(sizeof(unsigned));
  float* hbuf           = (float*)take(HID * sizeof(float));
  float* rhbuf          = (float*)take(HID * sizeof(float));
  __hip_bfloat16* Xbf   = (__hip_bfloat16*)take((size_t)T_FRAMES * HID * 2);
  float* Y              = (float*)take((size_t)T_FRAMES * HID * 4);
  float* Gz             = (float*)take((size_t)T_FRAMES * HID * 4);
  float* Gr             = (float*)take((size_t)T_FRAMES * HID * 4);
  float* Gh             = (float*)take((size_t)T_FRAMES * HID * 4);
  __hip_bfloat16* WzBf  = (__hip_bfloat16*)take((size_t)NLAYER * HID * HID * 2);
  __hip_bfloat16* WrBf  = (__hip_bfloat16*)take((size_t)NLAYER * HID * HID * 2);
  __hip_bfloat16* WhBf  = (__hip_bfloat16*)take((size_t)NLAYER * HID * HID * 2);
  __hip_bfloat16* fcWbf = (__hip_bfloat16*)take((size_t)OUTF * HID * 2);

  init_counter<<<1, 1, 0, stream>>>(counter);

  const int CB = 256, CG = 1024;
  cvt_f32_bf16<<<CG, CB, 0, stream>>>(x, Xbf, T_FRAMES * HID);
  cvt_f32_bf16<<<CG, CB, 0, stream>>>(Wz, WzBf, NLAYER * HID * HID);
  cvt_f32_bf16<<<CG, CB, 0, stream>>>(Wr, WrBf, NLAYER * HID * HID);
  cvt_f32_bf16<<<CG, CB, 0, stream>>>(Wh, WhBf, NLAYER * HID * HID);
  cvt_f32_bf16<<<CG, CB, 0, stream>>>(fcW, fcWbf, OUTF * HID);

  const size_t SMEM =
      (size_t)(3 * RPW * HID + 2 * HID + 2 * RPW + 3 * RPW) * sizeof(float);
  (void)hipFuncSetAttribute((const void*)gru_layer_recurrence,
                            hipFuncAttributeMaxDynamicSharedMemorySize,
                            (int)SMEM);   // ~197KB dynamic LDS (gfx1250: 320KB)

  const dim3 blk(256);
  const dim3 gridH(T_FRAMES / 128, HID / 64);
  for (int l = 0; l < NLAYER; ++l) {
    const size_t wo = (size_t)l * HID * HID;
    wmma_gemm_xwT<<<gridH, blk, 0, stream>>>(Xbf, WzBf + wo, nullptr, Gz,
                                             T_FRAMES, HID, HID);
    wmma_gemm_xwT<<<gridH, blk, 0, stream>>>(Xbf, WrBf + wo, nullptr, Gr,
                                             T_FRAMES, HID, HID);
    wmma_gemm_xwT<<<gridH, blk, 0, stream>>>(Xbf, WhBf + wo, nullptr, Gh,
                                             T_FRAMES, HID, HID);
    gru_layer_recurrence<<<NWG, blk, SMEM, stream>>>(
        Uz + wo, Ur + wo, Uh + wo, Gz, Gr, Gh,
        bWz + l * HID, bUz + l * HID, bWr + l * HID, bUr + l * HID,
        bWh + l * HID, bUh + l * HID, hbuf, rhbuf, counter, Y,
        out + (size_t)T_FRAMES * OUTF + (size_t)l * HID, l, T_FRAMES);
    cvt_f32_bf16<<<CG, CB, 0, stream>>>(Y, Xbf, T_FRAMES * HID);
  }

  const dim3 gridO(T_FRAMES / 128, OUTF / 64);
  wmma_gemm_xwT<<<gridO, blk, 0, stream>>>(Xbf, fcWbf, fcb, out, T_FRAMES,
                                           OUTF, HID);
}
